// Encoder_53025666236986
// MI455X (gfx1250) — compile-verified, matrix operand order checked by reading, not run on hardware
//
#include <hip/hip_runtime.h>
#include <hip/hip_bf16.h>

typedef __attribute__((ext_vector_type(16))) _Float16 v16h;
typedef __attribute__((ext_vector_type(8)))  _Float16 v8h;
typedef __attribute__((ext_vector_type(8)))  float    v8f;
typedef __attribute__((ext_vector_type(2)))  float    v2f;

#define B_TOT 65536
#define T_SEQ 32
#define E_DIM 20
#define H_DIM 100
#define FH    400           // 4*H
#define Z_DIM 100
#define V_SZ  50
#define ROWS  32            // batch rows per workgroup
#define NTHREADS 320        // 10 wave32 waves
#define NWAVES 10

#define H16_STRIDE 136      // halves per h16 row (128 data + pad, bank-stagger)
#define H32_STRIDE 104      // floats per h32/state/zm row
#define W_STRIDE   112      // floats per weight-tile row (100 + zero pad)
#define Z_STRIDE   400

// workspace layout (bytes)
#define TP_OFF   0          // 20000 f32  (token projection table)
#define RECB_OFF 80000      // 51200 f16  (rec weights pre-swizzled into B-fragment order)

// LDS layout (bytes). Phase-2 (dense head) overlays the tp region.
#define LDS_WBUF  0         // 44800: one 100x112 f32 weight tile
#define LDS_STATE 44800     // 13312
#define LDS_ZM    58112     // 13312
#define LDS_Z     80000     // 51200: z pre-activations 32x400 f32
#define LDS_H16   131200    // 8704 : h as f16, K padded to 128 (WMMA A source)
#define LDS_H32   139904    // 13312: running h in f32 (exact head input)
#define LDS_TOK   153216    // 4096 : all 32x32 tokens for this block
#define LDS_BYTES 157312

// ---------------- helpers ----------------

// Async global->LDS 16B copy (GLOBAL_LOAD_ASYNC_TO_LDS_B128, tracked by ASYNCcnt).
// LDS byte address = low 32 bits of the generic pointer (flat LDS aperture rule).
__device__ __forceinline__ void async_g2l_b128(void* lds_dst, const void* gsrc) {
  uint32_t loff = (uint32_t)(uintptr_t)lds_dst;
  asm volatile("global_load_async_to_lds_b128 %0, %1, off"
               :: "v"(loff), "v"(gsrc) : "memory");
}
__device__ __forceinline__ void async_wait0() {
  asm volatile("s_wait_asynccnt 0x0" ::: "memory");
}

__device__ __forceinline__ float fast_sigmoid(float x) {
  return __builtin_amdgcn_rcpf(1.f + __expf(-x));
}

// ---------------- prep kernels ----------------

// tp[v][j] = bias[j] + sum_e embed[v][e]*kernel[e][j]  (folds embedding+input GEMM+bias)
__global__ void tp_kernel(const float* __restrict__ embed, const float* __restrict__ kern,
                          const float* __restrict__ bias, float* __restrict__ tp) {
  int idx = blockIdx.x * 256 + threadIdx.x;
  if (idx >= V_SZ * FH) return;
  int v = idx / FH, j = idx % FH;
  float s = bias[j];
#pragma unroll
  for (int e = 0; e < E_DIM; e++) s += embed[v * E_DIM + e] * kern[e * FH + j];
  tp[idx] = s;
}

// Pre-swizzle rec (100x400 f32) into per-lane v16h B fragments for
// V_WMMA_F32_16X16X32_F16: index = ((nt*4+kc)*32 + lane)*16 + i
// lane<16 holds col N=nt*16+lane, K = {0..7,16..23}+kc*32 ; lane>=16 holds K={8..15,24..31}
__global__ void recb_kernel(const float* __restrict__ rec, _Float16* __restrict__ recB) {
  int idx = blockIdx.x * 256 + threadIdx.x;
  if (idx >= 25 * 4 * 32 * 16) return;
  int i = idx & 15, lane = (idx >> 4) & 31, kc = (idx >> 9) & 3, nt = idx >> 11;
  int kl = ((lane < 16) ? 0 : 8) + (i & 7) + ((i >> 3) << 4);
  int k = kc * 32 + kl;
  int n = nt * 16 + (lane & 15);
  float v = (k < H_DIM) ? rec[k * FH + n] : 0.f;
  recB[idx] = (_Float16)v;
}

// ---------------- exact f32 16x16x4 tile GEMM for the dense head ----------------
__device__ __forceinline__ v8f gemm_f32_tile(const float* __restrict__ A, int lda,
                                             const float* __restrict__ Bw,
                                             int mbase, int n0, int lane) {
  v8f acc = {};
  int mm = mbase + (lane & 15);
  int nn = n0 + (lane & 15);
  int kadd = (lane < 16) ? 0 : 2;
#if __has_builtin(__builtin_amdgcn_wmma_f32_16x16x4_f32)
#pragma unroll 5
  for (int kc = 0; kc < 25; kc++) {
    int k0 = kc * 4 + kadd;
    v2f a; a.x = A[mm * lda + k0];       a.y = A[mm * lda + k0 + 1];
    v2f b; b.x = Bw[k0 * W_STRIDE + nn]; b.y = Bw[(k0 + 1) * W_STRIDE + nn];
    acc = __builtin_amdgcn_wmma_f32_16x16x4_f32(false, a, false, b, (short)0, acc,
                                                false, false);
  }
#else
  int rb = (lane < 16) ? 0 : 8;
#pragma unroll
  for (int r = 0; r < 8; r++) {
    float s = 0.f;
    for (int k = 0; k < H_DIM; k++)
      s += A[(mbase + rb + r) * lda + k] * Bw[k * W_STRIDE + nn];
    acc[r] = s;
  }
  (void)mm; (void)kadd;
#endif
  return acc;
}

// ---------------- main fused encoder ----------------
__global__ __launch_bounds__(NTHREADS, 1)
void encoder_kernel(const int* __restrict__ data,
                    const float* __restrict__ tp_g,
                    const _Float16* __restrict__ recB,
                    const float* __restrict__ fc1_W, const float* __restrict__ fc1_b,
                    const float* __restrict__ mean_W, const float* __restrict__ mean_b,
                    const float* __restrict__ var_W, const float* __restrict__ var_b,
                    const float* __restrict__ eps,
                    float* __restrict__ out) {
  extern __shared__ char smem[];
  float*    tp      = (float*)(smem);              // phase 1
  float*    zbuf    = (float*)(smem + LDS_Z);
  _Float16* h16     = (_Float16*)(smem + LDS_H16);
  float*    h32     = (float*)(smem + LDS_H32);
  int*      tok_all = (int*)(smem + LDS_TOK);
  float*    wbuf    = (float*)(smem + LDS_WBUF);   // phase 2 (overlays tp)
  float*    st      = (float*)(smem + LDS_STATE);
  float*    zm      = (float*)(smem + LDS_ZM);

  const int tid   = threadIdx.x;
  const int lane  = tid & 31;
  const int wv    = tid >> 5;
  const int mt    = wv / 5;       // which 16-row M tile (0..1)
  const int strip = wv % 5;       // which 5-wide N strip (0..4)
  const int b0    = blockIdx.x * ROWS;

  // ---- async stage: tp table (80KB) + this block's 32x32 token tile (4KB) ----
  for (int i = tid; i < (V_SZ * FH) / 4; i += NTHREADS)
    async_g2l_b128(tp + i * 4, tp_g + i * 4);
  if (tid < (ROWS * T_SEQ) / 4)
    async_g2l_b128(tok_all + tid * 4, data + b0 * T_SEQ + tid * 4);

  // zero h buffers (f16 K-pad must stay zero) while async copies fly
  for (int i = tid; i < ROWS * H16_STRIDE; i += NTHREADS) h16[i] = (_Float16)0.f;
  for (int i = tid; i < ROWS * H32_STRIDE; i += NTHREADS) h32[i] = 0.f;

  // recurrent weights: B fragments cached in VGPRs for all 32 steps
  v16h Bf[5][4];
#pragma unroll
  for (int s = 0; s < 5; s++) {
    int nt = strip * 5 + s;
#pragma unroll
    for (int kc = 0; kc < 4; kc++)
      Bf[s][kc] = *(const v16h*)(recB + (((nt * 4 + kc) * 32 + lane) << 4));
  }

  // elementwise ownership: 10 (row, hcol) cells per thread; c stays in registers
  int er[10], ej[10];
  float c[10];
#pragma unroll
  for (int e = 0; e < 10; e++) {
    int idx = tid + e * NTHREADS;
    er[e] = idx / H_DIM; ej[e] = idx % H_DIM; c[e] = 0.f;
  }

  const int m      = mt * 16 + (lane & 15);
  const int koff   = (lane < 16) ? 0 : 8;
  const int rShift = (lane < 16) ? 0 : 8;

  async_wait0();   // tp + tokens resident before first barrier

  for (int t = 0; t < T_SEQ; t++) {
    __syncthreads();   // prev-step h16 writes & z consumers done

    // A fragments: h16 row m, K chunk layout per 16-bit A spec
    v16h Af[4];
#pragma unroll
    for (int kc = 0; kc < 4; kc++) {
      const _Float16* hp = h16 + m * H16_STRIDE + kc * 32 + koff;
      v8h lo = *(const v8h*)hp;
      v8h hi = *(const v8h*)(hp + 16);
      v16h a;
#pragma unroll
      for (int q = 0; q < 8; q++) { a[q] = lo[q]; a[q + 8] = hi[q]; }
      Af[kc] = a;
    }

    // z[m-tile, strip] = h @ rec   (5 N-tiles x 4 chained K chunks)
#pragma unroll
    for (int s = 0; s < 5; s++) {
      v8f acc = {};
#pragma unroll
      for (int kc = 0; kc < 4; kc++)
        acc = __builtin_amdgcn_wmma_f32_16x16x32_f16(false, Af[kc], false, Bf[s][kc],
                                                     (short)0, acc, false, false);
      int col   = (strip * 5 + s) * 16 + (lane & 15);
      int rbase = mt * 16 + rShift;
#pragma unroll
      for (int r = 0; r < 8; r++)
        zbuf[(rbase + r) * Z_STRIDE + col] = acc[r];
    }

    __syncthreads();

    // LSTM cell: gates = z + tp[token]; masked update
#pragma unroll
    for (int e = 0; e < 10; e++) {
      int r = er[e], j = ej[e];
      int tk = tok_all[r * T_SEQ + t];
      const float* tpr = tp + tk * FH;
      const float* zr  = zbuf + r * Z_STRIDE;
      float zi = zr[j]       + tpr[j];
      float zf = zr[j + 100] + tpr[j + 100];
      float zg = zr[j + 200] + tpr[j + 200];
      float zo = zr[j + 300] + tpr[j + 300];
      float ig = fast_sigmoid(zi);
      float fg = fast_sigmoid(zf);
      float gg = fmaxf(zg, 0.f);
      float og = fast_sigmoid(zo);
      float cn = fg * c[e] + ig * gg;
      float hn = og * fmaxf(cn, 0.f);
      if (tk != 0) {
        c[e] = cn;
        h32[r * H32_STRIDE + j] = hn;
        h16[r * H16_STRIDE + j] = (_Float16)hn;
      }
    }
  }

  __syncthreads();

  // ---------------- dense head, exact f32 WMMA ----------------
  const int SEC = B_TOT * Z_DIM;

  // 1) state = sigmoid(h @ fc1 + b)
  for (int i4 = tid; i4 < H_DIM * (W_STRIDE / 4); i4 += NTHREADS) {
    int k = i4 / (W_STRIDE / 4), nq = i4 % (W_STRIDE / 4);
    float4 v = (nq < 25) ? *(const float4*)(fc1_W + k * H_DIM + nq * 4)
                         : make_float4(0.f, 0.f, 0.f, 0.f);
    *(float4*)(wbuf + k * W_STRIDE + nq * 4) = v;
  }
  __syncthreads();
  for (int job = wv; job < 14; job += NWAVES) {
    int jm = job / 7, n0 = (job % 7) * 16;
    v8f acc = gemm_f32_tile(h32, H32_STRIDE, wbuf, jm * 16, n0, lane);
    int nn = n0 + (lane & 15);
    if (nn < H_DIM) {
      float bv = fc1_b[nn];
      int rbase = jm * 16 + rShift;
#pragma unroll
      for (int r = 0; r < 8; r++)
        st[(rbase + r) * H32_STRIDE + nn] = fast_sigmoid(acc[r] + bv);
    }
  }
  __syncthreads();

  // 2) z_mean = state @ mean_W + mean_b
  for (int i4 = tid; i4 < H_DIM * (W_STRIDE / 4); i4 += NTHREADS) {
    int k = i4 / (W_STRIDE / 4), nq = i4 % (W_STRIDE / 4);
    float4 v = (nq < 25) ? *(const float4*)(mean_W + k * H_DIM + nq * 4)
                         : make_float4(0.f, 0.f, 0.f, 0.f);
    *(float4*)(wbuf + k * W_STRIDE + nq * 4) = v;
  }
  __syncthreads();
  for (int job = wv; job < 14; job += NWAVES) {
    int jm = job / 7, n0 = (job % 7) * 16;
    v8f acc = gemm_f32_tile(st, H32_STRIDE, wbuf, jm * 16, n0, lane);
    int nn = n0 + (lane & 15);
    if (nn < H_DIM) {
      float bv = mean_b[nn];
      int rbase = jm * 16 + rShift;
#pragma unroll
      for (int r = 0; r < 8; r++) {
        float v = acc[r] + bv;
        zm[(rbase + r) * H32_STRIDE + nn] = v;
        out[(b0 + rbase + r) * Z_DIM + nn] = v;
      }
    }
  }
  __syncthreads();

  // 3) z_log_var = state @ var_W + var_b ; z = z_mean + exp(0.5*zlv)*eps
  for (int i4 = tid; i4 < H_DIM * (W_STRIDE / 4); i4 += NTHREADS) {
    int k = i4 / (W_STRIDE / 4), nq = i4 % (W_STRIDE / 4);
    float4 v = (nq < 25) ? *(const float4*)(var_W + k * H_DIM + nq * 4)
                         : make_float4(0.f, 0.f, 0.f, 0.f);
    *(float4*)(wbuf + k * W_STRIDE + nq * 4) = v;
  }
  __syncthreads();
  for (int job = wv; job < 14; job += NWAVES) {
    int jm = job / 7, n0 = (job % 7) * 16;
    v8f acc = gemm_f32_tile(st, H32_STRIDE, wbuf, jm * 16, n0, lane);
    int nn = n0 + (lane & 15);
    if (nn < H_DIM) {
      float bv = var_b[nn];
      int rbase = jm * 16 + rShift;
#pragma unroll
      for (int r = 0; r < 8; r++) {
        int g = b0 + rbase + r;
        float zlv = acc[r] + bv;
        out[SEC + g * Z_DIM + nn] = zlv;
        out[2 * SEC + g * Z_DIM + nn] =
            zm[(rbase + r) * H32_STRIDE + nn] + __expf(0.5f * zlv) * eps[g * Z_DIM + nn];
      }
    }
  }
}

extern "C" void kernel_launch(void* const* d_in, const int* in_sizes, int n_in,
                              void* d_out, int out_size, void* d_ws, size_t ws_size,
                              hipStream_t stream) {
  const int*   data  = (const int*)d_in[0];
  const float* embed = (const float*)d_in[1];
  const float* kern  = (const float*)d_in[2];
  const float* rec   = (const float*)d_in[3];
  const float* bias  = (const float*)d_in[4];
  const float* fc1W  = (const float*)d_in[5];
  const float* fc1b  = (const float*)d_in[6];
  const float* meanW = (const float*)d_in[7];
  const float* meanb = (const float*)d_in[8];
  const float* varW  = (const float*)d_in[9];
  const float* varb  = (const float*)d_in[10];
  const float* eps   = (const float*)d_in[11];
  float* out = (float*)d_out;

  float*    tp   = (float*)((char*)d_ws + TP_OFF);
  _Float16* recB = (_Float16*)((char*)d_ws + RECB_OFF);

  tp_kernel<<<(V_SZ * FH + 255) / 256, 256, 0, stream>>>(embed, kern, bias, tp);
  recb_kernel<<<(25 * 4 * 32 * 16 + 255) / 256, 256, 0, stream>>>(rec, recB);
  encoder_kernel<<<B_TOT / ROWS, NTHREADS, LDS_BYTES, stream>>>(
      data, tp, recB, fc1W, fc1b, meanW, meanb, varW, varb, eps, out);
}
